// DeformableExpressionAttention_39462159516114
// MI455X (gfx1250) — matrix operand order repaired and checked by reading, not compiled, not run on hardware
//
#include <hip/hip_runtime.h>
#include <hip/hip_bf16.h>
#include <math.h>

// CDNA5 WMMA vector types (probe-confirmed signatures)
typedef __attribute__((ext_vector_type(16))) __bf16 v16bf;
typedef __attribute__((ext_vector_type(2)))  __bf16 v2bf;
typedef __attribute__((ext_vector_type(8)))  float  v8f;

namespace {
constexpr int Bb  = 4;
constexpr int Ll  = 4096;
constexpr int Dd  = 1024;
constexpr int Hh  = 8;
constexpr int Pp  = 4;
constexpr int DPH = Dd / Hh;   // 128
constexpr int KSTEP = 32;      // K per WMMA step
constexpr int BPITCH = 40;     // LDS column pitch in bf16 units (80B):
                               // 16B-aligned fragments, 20*l mod 64 bank spread
}

struct SampleDesc { int i0; int i1; float w0; float w1; };

struct BF16x16 { uint4 lo; uint4 hi; };   // 32B carrier for a v16bf fragment

static __device__ __forceinline__ unsigned pack2bf(float a, float b) {
  v2bf t;
  t[0] = (__bf16)a;
  t[1] = (__bf16)b;
  return __builtin_bit_cast(unsigned, t);  // single v_cvt_pk_bf16_f32
}

// ---------------------------------------------------------------------------
// GEMM: C[M,N] = A[M,K] @ W[K,N] + bias[N]
// f32 in/out, bf16 WMMA compute (v_wmma_f32_16x16x32_bf16).
// Block = 256 threads = 8 waves. Wave tile = 32(M) x 64(N) -> 8 WMMAs/K-step.
// Block tile = 256(M) x 64(N).
//  * B chunk (32x64) staged in LDS as bf16, column-major, DOUBLE-BUFFERED:
//    one barrier per K-step; the convert+ds_store of chunk k+1 and its
//    register-staged global fetch overlap the WMMA stream of chunk k.
//  * Fragments read back as two aligned ds_load_b128, zero per-read cvts.
// M must be a multiple of 256; N guarded at staging/epilogue (handles 32/64).
// ---------------------------------------------------------------------------
__global__ __launch_bounds__(256) void wmma_gemm_bias_kernel(
    const float* __restrict__ A, const float* __restrict__ W,
    const float* __restrict__ bias, float* __restrict__ C,
    int N, int K)
{
  __shared__ __align__(16) unsigned short Bs[2][64 * BPITCH];   // 2 x 5 KB

  const int tid   = threadIdx.x;
  const int lane  = tid & 31;
  const int wave  = tid >> 5;
  const int lg    = lane >> 4;       // lane group (0: lanes 0-15, 1: 16-31)
  const int lr    = lane & 15;       // row (A) / col (B,C) within a 16-tile
  const int mBase = blockIdx.y * 256 + wave * 32;
  const int nBase = blockIdx.x * 64;

  // staging map: thread covers K-rows {krow0, krow0+1}, 4 cols each
  const int krow0 = (tid >> 4) * 2;  // 0,2,...,30
  const int scol  = (tid & 15) * 4;  // 0..60  (N is always a multiple of 4)
  const bool sOK  = (nBase + scol) < N;
  const float* __restrict__ wbase = W + nBase + scol;

  v8f acc[2][4] = {};

  const float* __restrict__ arow0 = A + (size_t)(mBase + lr) * K;
  const float* __restrict__ arow1 = arow0 + (size_t)16 * K;

  float4 wreg0 = make_float4(0.f, 0.f, 0.f, 0.f);
  float4 wreg1 = wreg0;

  // convert + commit the register-staged chunk into LDS buffer `buf`
  auto storeStage = [&](int buf) {
    unsigned short* base = &Bs[buf][0];
    *(unsigned*)&base[(scol + 0) * BPITCH + krow0] = pack2bf(wreg0.x, wreg1.x);
    *(unsigned*)&base[(scol + 1) * BPITCH + krow0] = pack2bf(wreg0.y, wreg1.y);
    *(unsigned*)&base[(scol + 2) * BPITCH + krow0] = pack2bf(wreg0.z, wreg1.z);
    *(unsigned*)&base[(scol + 3) * BPITCH + krow0] = pack2bf(wreg0.w, wreg1.w);
  };
  auto fetchW = [&](int kk) {
    wreg0 = *(const float4*)(wbase + (size_t)(kk + krow0) * N);
    wreg1 = *(const float4*)(wbase + (size_t)(kk + krow0 + 1) * N);
  };

  // --- prologue: stage chunk 0 into buffer 0 ---
  if (sOK) fetchW(0);
  storeStage(0);
  __syncthreads();

  int buf = 0;
  for (int kk = 0; kk < K; kk += KSTEP) {
    const bool hasNext = (kk + KSTEP) < K;

    // issue next chunk's global fetch early (latency hidden under WMMAs)
    if (hasNext && sOK) fetchW(kk + KSTEP);

    // ---- A fragments (k = 16*lg + s, matching B): contiguous 32B chunks ----
    const int kA = kk + lg * 16;
    v16bf af0, af1;
#pragma unroll
    for (int i = 0; i < 8; ++i) {
      af0[i]     = (__bf16)arow0[kA + i];
      af0[i + 8] = (__bf16)arow0[kA + 8 + i];
      af1[i]     = (__bf16)arow1[kA + i];
      af1[i + 8] = (__bf16)arow1[kA + 8 + i];
    }
    __builtin_prefetch(arow0 + kk + KSTEP, 0, 1);
    __builtin_prefetch(arow1 + kk + KSTEP, 0, 1);

    // ---- B fragments from LDS buffer `buf` (2 x ds_load_b128) + 8 WMMAs --
    // B k order, matching A: lanes 0-15 hold K = kk+0..15, lanes 16-31 hold +16.
#pragma unroll
    for (int t = 0; t < 4; ++t) {
      const unsigned short* bp = &Bs[buf][(t * 16 + lr) * BPITCH + lg * 16];
      BF16x16 raw;
      raw.lo = *(const uint4*)(bp);       // 16B aligned
      raw.hi = *(const uint4*)(bp + 8);   // +16B
      const v16bf bf = __builtin_bit_cast(v16bf, raw);
      acc[0][t] = __builtin_amdgcn_wmma_f32_16x16x32_bf16(
          false, af0, false, bf, (short)0, acc[0][t], false, false);
      acc[1][t] = __builtin_amdgcn_wmma_f32_16x16x32_bf16(
          false, af1, false, bf, (short)0, acc[1][t], false, false);
    }

    // ---- stage next chunk into the other buffer; single barrier ----
    if (hasNext) storeStage(buf ^ 1);
    __syncthreads();
    buf ^= 1;
  }

  // ---- epilogue: C layout: VGPR r -> row r + 8*laneGroup, col = lr ----
#pragma unroll
  for (int t = 0; t < 4; ++t) {
    const int col = nBase + t * 16 + lr;
    if (col >= N) continue;
    const float bval = bias[col];
#pragma unroll
    for (int s = 0; s < 2; ++s) {
#pragma unroll
      for (int r = 0; r < 8; ++r) {
        const int row = mBase + s * 16 + r + 8 * lg;
        C[(size_t)row * N + col] = acc[s][t][r] + bval;
      }
    }
  }
}

// ---------------------------------------------------------------------------
// Fuse softmax(P=4), ref+offset clip, grid_sample coordinate math, boundary
// masks and the height-1 row weight into per-(b,l,h,p) sample descriptors:
//   contribution = w0 * v[i0] + w1 * v[i1]
// ---------------------------------------------------------------------------
__global__ __launch_bounds__(256) void prep_kernel(
    const float* __restrict__ off_raw,   // (B,L,H,P,2)
    const float* __restrict__ attn_raw,  // (B,L,H,P)
    SampleDesc* __restrict__ desc)       // (B,L,H,P)
{
  const int g = blockIdx.x * blockDim.x + threadIdx.x;  // over B*L*H
  if (g >= Bb * Ll * Hh) return;
  const int l = (g / Hh) % Ll;

  const float* ar = attn_raw + (size_t)g * Pp;
  const float a0 = ar[0], a1 = ar[1], a2 = ar[2], a3 = ar[3];
  const float mx = fmaxf(fmaxf(a0, a1), fmaxf(a2, a3));
  const float e0 = __expf(a0 - mx), e1 = __expf(a1 - mx),
              e2 = __expf(a2 - mx), e3 = __expf(a3 - mx);
  const float inv = 1.0f / (e0 + e1 + e2 + e3);
  const float aw[4] = { e0 * inv, e1 * inv, e2 * inv, e3 * inv };

  const float ref_y = (float)l * (1.0f / (float)(Ll - 1));
  const float* orow = off_raw + (size_t)g * Pp * 2;

#pragma unroll
  for (int p = 0; p < Pp; ++p) {
    const float x = fminf(fmaxf(orow[2 * p + 0], 0.0f), 1.0f);          // ref_x = 0
    const float y = fminf(fmaxf(ref_y + orow[2 * p + 1], 0.0f), 1.0f);
    // unnormalize, align_corners=False: u = ((c+1)*size - 1)/2
    const float ux = ((x + 1.0f) * (float)Ll - 1.0f) * 0.5f;            // width = L
    const float uy = y * 0.5f;                                          // height = 1
    const float x0f = floorf(ux);
    const float wx1 = ux - x0f, wx0 = 1.0f - wx1;
    const float y0f = floorf(uy);
    const float wy1 = uy - y0f, wy0 = 1.0f - wy1;
    const float wy  = wy0 * (y0f == 0.0f ? 1.0f : 0.0f)
                    + wy1 * ((y0f + 1.0f) == 0.0f ? 1.0f : 0.0f);
    const int   i0  = (int)x0f, i1 = i0 + 1;
    const float m0  = (i0 >= 0 && i0 < Ll) ? 1.0f : 0.0f;
    const float m1  = (i1 >= 0 && i1 < Ll) ? 1.0f : 0.0f;
    SampleDesc d;
    d.i0 = min(max(i0, 0), Ll - 1);
    d.i1 = min(max(i1, 0), Ll - 1);
    const float s = aw[p] * wy;
    d.w0 = wx0 * m0 * s;
    d.w1 = wx1 * m1 * s;
    desc[(size_t)g * Pp + p] = d;
  }
}

// ---------------------------------------------------------------------------
// Gather/blend: mid[b,l,h*DPH+c] = sum_p w0*v[b,i0,hc] + w1*v[b,i1,hc]
// Threads consecutive in channel -> v gathers are fully coalesced per tap.
// ---------------------------------------------------------------------------
__global__ __launch_bounds__(256) void sample_kernel(
    const float* __restrict__ v, const SampleDesc* __restrict__ desc,
    float* __restrict__ mid)
{
  const size_t idx = (size_t)blockIdx.x * blockDim.x + threadIdx.x;  // B*L*D
  if (idx >= (size_t)Bb * Ll * Dd) return;
  const int    cd = (int)(idx % Dd);     // channel within D (== h*DPH + c)
  const size_t bl = idx / Dd;            // b*L + l
  const int    h  = cd / DPH;
  const size_t b  = bl / Ll;

  const SampleDesc* dp = desc + (bl * Hh + h) * Pp;
  const float* vb = v + (b * Ll) * (size_t)Dd + cd;

  float acc = 0.0f;
#pragma unroll
  for (int p = 0; p < Pp; ++p) {
    const SampleDesc d = dp[p];
    acc += d.w0 * vb[(size_t)d.i0 * Dd] + d.w1 * vb[(size_t)d.i1 * Dd];
  }
  mid[idx] = acc;
}

// ---------------------------------------------------------------------------
extern "C" void kernel_launch(void* const* d_in, const int* in_sizes, int n_in,
                              void* d_out, int out_size, void* d_ws, size_t ws_size,
                              hipStream_t stream) {
  (void)in_sizes; (void)n_in; (void)out_size; (void)ws_size;

  // Inputs (setup_inputs order). key_in/Wqk/bqk are dead code in the reference.
  const float* query = (const float*)d_in[0];
  const float* value = (const float*)d_in[2];
  const float* Wv    = (const float*)d_in[5];
  const float* bv    = (const float*)d_in[6];
  const float* Woff  = (const float*)d_in[7];
  const float* boff  = (const float*)d_in[8];
  const float* Wattn = (const float*)d_in[9];
  const float* battn = (const float*)d_in[10];
  const float* Wo    = (const float*)d_in[11];
  const float* bo    = (const float*)d_in[12];
  float* out = (float*)d_out;

  const int M = Bb * Ll;                 // 16384

  // Workspace carve-up (~142 MB)
  float*      v_ws     = (float*)d_ws;                               // M*D
  float*      off_raw  = v_ws + (size_t)M * Dd;                      // M*H*P*2
  float*      attn_raw = off_raw + (size_t)M * Hh * Pp * 2;          // M*H*P
  SampleDesc* desc     = (SampleDesc*)(attn_raw + (size_t)M * Hh * Pp);
  float*      mid      = (float*)(desc + (size_t)M * Hh * Pp);       // M*D

  dim3 blk(256);

  // v = value @ Wv + bv
  wmma_gemm_bias_kernel<<<dim3(Dd / 64, M / 256), blk, 0, stream>>>(
      value, Wv, bv, v_ws, Dd, Dd);
  // offset logits = query @ Woff + boff  (N = 64)
  wmma_gemm_bias_kernel<<<dim3(1, M / 256), blk, 0, stream>>>(
      query, Woff, boff, off_raw, Hh * Pp * 2, Dd);
  // attn logits = query @ Wattn + battn  (N = 32)
  wmma_gemm_bias_kernel<<<dim3(1, M / 256), blk, 0, stream>>>(
      query, Wattn, battn, attn_raw, Hh * Pp, Dd);
  // softmax + clip + bilinear setup -> descriptors
  prep_kernel<<<dim3((Bb * Ll * Hh + 255) / 256), blk, 0, stream>>>(
      off_raw, attn_raw, desc);
  // gather/blend -> mid
  sample_kernel<<<dim3((unsigned)(((size_t)M * Dd + 255) / 256)), blk, 0, stream>>>(
      v_ws, desc, mid);
  // out = mid @ Wo + bo
  wmma_gemm_bias_kernel<<<dim3(Dd / 64, M / 256), blk, 0, stream>>>(
      mid, Wo, bo, out, Dd, Dd);
}
